// MultiHeadAttention_24309514895753
// MI455X (gfx1250) — matrix operand order repaired and checked by reading, not compiled, not run on hardware
//
#include <hip/hip_runtime.h>
#include <hip/hip_bf16.h>

typedef __attribute__((ext_vector_type(16))) __bf16 v16bf;
typedef __attribute__((ext_vector_type(8)))  float  v8f;

#define DEV static __device__ __forceinline__

// ---------------- WMMA fragment helpers (bf16 16x16x32, wave32) ----------------
// A matrix 16x32: lane m = lane&15; half h = lane>>4.
// elements e=0..7  -> K = 8h + e        (contiguous)
// elements e=8..15 -> K = 16 + 8h + (e-8) (contiguous)
DEV v16bf load_a_f32(const float* base, int ld, int lane, int k0, float scale) {
  int m = lane & 15, h = (lane >> 4) & 1;
  const float* p = base + (size_t)m * ld + k0 + 8 * h;
  v16bf a;
#pragma unroll
  for (int e = 0; e < 8; ++e) {
    a[e]     = (__bf16)(p[e] * scale);
    a[e + 8] = (__bf16)(p[e + 16] * scale);
  }
  return a;
}

DEV v16bf load_a_bf16(const __bf16* base, int ld, int lane, int k0) {
  int m = lane & 15, h = (lane >> 4) & 1;
  const __bf16* p = base + (size_t)m * ld + k0 + 8 * h;
  v16bf a;
#pragma unroll
  for (int e = 0; e < 8; ++e) { a[e] = p[e]; a[e + 8] = p[e + 16]; }
  return a;
}

DEV v16bf load_a_lds(const float* S, int lane, int k0) {
  int m = lane & 15, h = (lane >> 4) & 1;
  const float* p = S + m * 1024 + k0 + 8 * h;
  v16bf a;
#pragma unroll
  for (int e = 0; e < 8; ++e) {
    a[e]     = (__bf16)p[e];
    a[e + 8] = (__bf16)p[e + 16];
  }
  return a;
}

// B matrix 32x16: lane n = lane&15, K = 8*half + e for e<8, 16 + 8*half + (e-8)
// for e>=8 (same as A). Source is row-major with rows indexed by N and columns by K
// (i.e. W[n][k], K[key][d], Vt[d][key]).
DEV v16bf load_b_f32(const float* W, int ld, int lane, int n0, int k0) {
  int n = lane & 15, h = (lane >> 4) & 1;
  const float* p = W + (size_t)(n0 + n) * ld + k0 + 8 * h;
  v16bf b;
#pragma unroll
  for (int e = 0; e < 8; ++e) { b[e] = (__bf16)p[e]; b[e + 8] = (__bf16)p[e + 16]; }
  return b;
}

DEV v16bf load_b_bf16(const __bf16* W, int ld, int lane, int n0, int k0) {
  int n = lane & 15, h = (lane >> 4) & 1;
  const __bf16* p = W + (size_t)(n0 + n) * ld + k0 + 8 * h;
  v16bf b;
#pragma unroll
  for (int e = 0; e < 8; ++e) { b[e] = p[e]; b[e + 8] = p[e + 16]; }
  return b;
}

DEV v8f wmma_bf16(v16bf a, v16bf b, v8f c) {
  return __builtin_amdgcn_wmma_f32_16x16x32_bf16(false, a, false, b, (short)0, c,
                                                 false, false);
}

// ---------------- Projection GEMM: out[M=8192][512] = X @ W^T (+bias) ----------
// mode 0: store bf16 row-major.  mode 1: store bf16 transposed per head
// (Vt[(b*8+h)*64+d][key]).  mode 2: store fp32 row-major (final output).
template <bool XBF>
__global__ __launch_bounds__(256) void proj_gemm(const void* Xv, const float* W,
                                                 const float* bias, __bf16* outBF,
                                                 float* outF, float scale,
                                                 int mode) {
  const int KD = 512, ND = 512;
  int lane = threadIdx.x & 31;
  int wave = (blockIdx.x << 3) + (threadIdx.x >> 5);
  int m0 = (wave >> 3) << 4;  // 512 row tiles of 16
  int n0 = (wave & 7) << 6;   // 8 column strips of 64
  const float*  Xf = (const float*)Xv;
  const __bf16* Xb = (const __bf16*)Xv;

  v8f c0 = {}, c1 = {}, c2 = {}, c3 = {};
  for (int k0 = 0; k0 < KD; k0 += 32) {
    v16bf a = XBF ? load_a_bf16(Xb + (size_t)m0 * KD, KD, lane, k0)
                  : load_a_f32(Xf + (size_t)m0 * KD, KD, lane, k0, scale);
    v16bf b0 = load_b_f32(W, KD, lane, n0,      k0);
    v16bf b1 = load_b_f32(W, KD, lane, n0 + 16, k0);
    v16bf b2 = load_b_f32(W, KD, lane, n0 + 32, k0);
    v16bf b3 = load_b_f32(W, KD, lane, n0 + 48, k0);
    c0 = wmma_bf16(a, b0, c0);
    c1 = wmma_bf16(a, b1, c1);
    c2 = wmma_bf16(a, b2, c2);
    c3 = wmma_bf16(a, b3, c3);
  }

  int n = lane & 15, h = (lane >> 4) & 1;
  v8f cs[4] = {c0, c1, c2, c3};
#pragma unroll
  for (int t = 0; t < 4; ++t) {
#pragma unroll
    for (int r = 0; r < 8; ++r) {
      int row = m0 + r + 8 * h;
      int col = n0 + t * 16 + n;
      float v = cs[t][r] + (bias ? bias[col] : 0.0f);
      if (mode == 2) {
        outF[(size_t)row * ND + col] = v;
      } else if (mode == 1) {
        int bidx = row >> 10, key = row & 1023, hh = col >> 6, d = col & 63;
        outBF[(((size_t)(bidx * 8 + hh) * 64 + d) << 10) + key] = (__bf16)v;
      } else {
        outBF[(size_t)row * ND + col] = (__bf16)v;
      }
    }
  }
}

// ---------------- Fused attention: scores -> sparsemax -> alpha @ V ------------
// One block per (16-query tile, head, batch). 256 threads = 8 waves.
__global__ __launch_bounds__(256) void attn_sparsemax(const __bf16* Qbf,
                                                      const __bf16* Kbf,
                                                      const __bf16* Vt,
                                                      __bf16* resbf) {
  __shared__ float S[16 * 1024];  // 64 KB score/alpha tile
  const int L = 1024, Dm = 512;
  int qt = blockIdx.x, hH = blockIdx.y, bB = blockIdx.z;
  int lane = threadIdx.x & 31, w = threadIdx.x >> 5;
  int n = lane & 15, hf = (lane >> 4) & 1;

  int m0 = bB * L + qt * 16;  // global query row base
  const __bf16* Qrow  = Qbf + (size_t)m0 * Dm + hH * 64;
  const __bf16* Khead = Kbf + (size_t)bB * L * Dm + hH * 64;

  // ---- Phase 1: S[16][1024] = (Q/temp) K^T. Wave w covers keys [w*128, w*128+128).
  v16bf aq0 = load_a_bf16(Qrow, Dm, lane, 0);
  v16bf aq1 = load_a_bf16(Qrow, Dm, lane, 32);
#pragma unroll
  for (int ct = 0; ct < 8; ++ct) {
    int key0 = (w * 8 + ct) * 16;
    v16bf b0 = load_b_bf16(Khead, Dm, lane, key0, 0);
    v16bf b1 = load_b_bf16(Khead, Dm, lane, key0, 32);
    v8f c = {};
    c = wmma_bf16(aq0, b0, c);
    c = wmma_bf16(aq1, b1, c);
#pragma unroll
    for (int r = 0; r < 8; ++r) S[(r + 8 * hf) * 1024 + key0 + n] = c[r];
  }
  __syncthreads();

  // ---- Phase 2: sparsemax per row (wave handles rows 2w, 2w+1).
  // tau solved by bisection on s(tau)=sum(max(z-tau,0)) in [max-1, max],
  // then exact closed form from the localized support set.
  for (int rr = 0; rr < 2; ++rr) {
    int row = w * 2 + rr;
    float* z = &S[row * 1024];
    float zr[32];
#pragma unroll
    for (int j = 0; j < 32; ++j) zr[j] = z[lane + 32 * j];
    float mx = -3.4e38f;
#pragma unroll
    for (int j = 0; j < 32; ++j) mx = fmaxf(mx, zr[j]);
    for (int o = 16; o; o >>= 1) mx = fmaxf(mx, __shfl_xor(mx, o, 32));
    float lo = mx - 1.0f, hi = mx;
    for (int it = 0; it < 24; ++it) {
      float mid = 0.5f * (lo + hi);
      float s = 0.0f;
#pragma unroll
      for (int j = 0; j < 32; ++j) {
        float d = zr[j] - mid;
        s += (d > 0.0f) ? d : 0.0f;
      }
      for (int o = 16; o; o >>= 1) s += __shfl_xor(s, o, 32);
      if (s >= 1.0f) lo = mid; else hi = mid;
    }
    float tau0 = 0.5f * (lo + hi);
    float cnt = 0.0f, sm = 0.0f;
#pragma unroll
    for (int j = 0; j < 32; ++j) {
      if (zr[j] > tau0) { cnt += 1.0f; sm += zr[j]; }
    }
    for (int o = 16; o; o >>= 1) {
      cnt += __shfl_xor(cnt, o, 32);
      sm  += __shfl_xor(sm, o, 32);
    }
    float tau = (sm - 1.0f) / cnt;
#pragma unroll
    for (int j = 0; j < 32; ++j) {
      float v = zr[j] - tau;
      z[lane + 32 * j] = (v > 0.0f) ? v : 0.0f;
    }
  }
  __syncthreads();

  // ---- Phase 3: res[16][64] = alpha[16][1024] @ V[1024][64].
  // Wave w -> output column tile t = w&3, K half kh = w>>2 (512 keys each).
  int t = w & 3, kh = w >> 2;
  const __bf16* Vhead = Vt + (size_t)(bB * 8 + hH) * 64 * 1024;  // [64][1024]
  v8f acc = {};
  for (int kc = kh * 16; kc < kh * 16 + 16; ++kc) {
    int k0 = kc * 32;
    v16bf a  = load_a_lds(S, lane, k0);
    v16bf bb = load_b_bf16(Vhead, 1024, lane, t * 16, k0);
    acc = wmma_bf16(a, bb, acc);
  }
  __syncthreads();  // all alpha reads done; safe to reuse S as staging
  float* stage = S;  // stage[4][16][16]
  if (kh == 1) {
#pragma unroll
    for (int r = 0; r < 8; ++r) stage[t * 256 + (r + 8 * hf) * 16 + n] = acc[r];
  }
  __syncthreads();
  if (kh == 0) {
#pragma unroll
    for (int r = 0; r < 8; ++r) {
      float v = acc[r] + stage[t * 256 + (r + 8 * hf) * 16 + n];
      int grow = m0 + r + 8 * hf;
      int col = hH * 64 + t * 16 + n;
      resbf[(size_t)grow * Dm + col] = (__bf16)v;
    }
  }
}

// ---------------- Host launch ---------------------------------------------------
extern "C" void kernel_launch(void* const* d_in, const int* in_sizes, int n_in,
                              void* d_out, int out_size, void* d_ws, size_t ws_size,
                              hipStream_t stream) {
  const float* h_q = (const float*)d_in[0];
  const float* h_k = (const float*)d_in[1];
  const float* h_v = (const float*)d_in[2];
  const float* Wq  = (const float*)d_in[3];
  const float* Wk  = (const float*)d_in[4];
  const float* Wv  = (const float*)d_in[5];
  const float* bv  = (const float*)d_in[6];
  const float* Wf  = (const float*)d_in[7];
  const float* bf  = (const float*)d_in[8];
  float* out = (float*)d_out;

  // workspace: Qbf | Kbf | Vt (per-head transposed) | resbf, all bf16 8192x512
  char* ws = (char*)d_ws;
  const size_t MAT = (size_t)8192 * 512;
  __bf16* Qbf   = (__bf16*)(ws);
  __bf16* Kbf   = (__bf16*)(ws + MAT * 2);
  __bf16* Vt    = (__bf16*)(ws + MAT * 4);
  __bf16* resbf = (__bf16*)(ws + MAT * 6);

  const float qscale = 0.125f;  // 1 / TEMPERATURE = 1 / sqrt(64)

  dim3 gblk(512), gthr(256);
  proj_gemm<false><<<gblk, gthr, 0, stream>>>(h_q, Wq, nullptr, Qbf, nullptr, qscale, 0);
  proj_gemm<false><<<gblk, gthr, 0, stream>>>(h_k, Wk, nullptr, Kbf, nullptr, 1.0f, 0);
  proj_gemm<false><<<gblk, gthr, 0, stream>>>(h_v, Wv, bv,      Vt,  nullptr, 1.0f, 1);
  attn_sparsemax<<<dim3(64, 8, 8), 256, 0, stream>>>(Qbf, Kbf, Vt, resbf);
  proj_gemm<true><<<gblk, gthr, 0, stream>>>(resbf, Wf, bf, nullptr, out, 1.0f, 2);
}
